// TransformerBlock_22119081574838
// MI455X (gfx1250) — matrix operand order repaired and checked by reading, not compiled, not run on hardware
//
#include <hip/hip_runtime.h>
#include <hip/hip_bf16.h>

// ---------------------------------------------------------------------------
// Transformer block (pre-LN, causal MHA + GELU FFN) for MI455X / gfx1250.
// All GEMMs + attention matmuls run on v_wmma_f32_16x16x32_bf16 (wave32).
// Global->LDS traffic uses gfx1250 async-to-LDS ops (ASYNCcnt) with
// double-buffered LDS tiles in the GEMM.
// Shapes fixed by the reference: B=4, S=2048, D=1024, H=16, Hd=64, Dff=4096.
// ---------------------------------------------------------------------------

typedef __attribute__((ext_vector_type(16))) __bf16 v16bf;
typedef __attribute__((ext_vector_type(8)))  float  v8f;

struct U4x2 { uint4 a, b; };

static __device__ __forceinline__ v16bf packfrag(uint4 lo, uint4 hi) {
    U4x2 t{lo, hi};
    return __builtin_bit_cast(v16bf, t);
}

static __device__ __forceinline__ v8f wmma_bf16(v16bf a, v16bf b, v8f c) {
    // (neg_a, A, neg_b, B, c_mod, C, reuse_a, reuse_b)
    return __builtin_amdgcn_wmma_f32_16x16x32_bf16(false, a, false, b,
                                                   (short)0, c, false, false);
}

// float -> bf16 bits, round-to-nearest-even
static __device__ __forceinline__ unsigned short f2bf(float f) {
    unsigned u = __builtin_bit_cast(unsigned, f);
    u += 0x7FFFu + ((u >> 16) & 1u);
    return (unsigned short)(u >> 16);
}

// Low 32 bits of a generic pointer to LDS = byte offset within wave's LDS
// (ISA 10.2: LDS aperture maps addr[31:0] to the LDS offset).
static __device__ __forceinline__ unsigned lds_off(const void* p) {
    return (unsigned)(size_t)p;
}

// gfx1250 async copy: 16B per lane, global -> LDS, tracked by ASYNCcnt.
// VDST = per-lane LDS byte offset, ADDR = 64-bit global address (GV mode).
static __device__ __forceinline__ void async_load_b128(unsigned ldsByteOff,
                                                       const void* gaddr) {
    asm volatile("global_load_async_to_lds_b128 %0, %1, off"
                 :
                 : "v"(ldsByteOff), "v"((unsigned long long)(size_t)gaddr)
                 : "memory");
}

static __device__ __forceinline__ void wait_async0() {
    asm volatile("s_wait_asynccnt 0x0" ::: "memory");
}

// ---------------------------------------------------------------------------
// Weight convert + transpose: W fp32 [K,N] row-major -> Wt bf16 [N,K]
// ---------------------------------------------------------------------------
__global__ __launch_bounds__(256) void wtrans_kern(const float* __restrict__ W,
                                                   unsigned short* __restrict__ Wt,
                                                   int K, int N) {
    int idx = blockIdx.x * 256 + threadIdx.x;
    if (idx < N * K) {
        int n = idx / K;
        int k = idx - n * K;
        Wt[idx] = f2bf(W[(size_t)k * N + n]);
    }
}

// ---------------------------------------------------------------------------
// LayerNorm (D=1024) fp32 -> bf16, one block per row, 256 threads
// ---------------------------------------------------------------------------
__global__ __launch_bounds__(256) void ln_kern(const float* __restrict__ x,
                                               const float* __restrict__ gamma,
                                               const float* __restrict__ beta,
                                               unsigned short* __restrict__ out) {
    __shared__ float rsum[256];
    __shared__ float rsq[256];
    const int row = blockIdx.x;
    const int t = threadIdx.x;
    const float4 v = ((const float4*)(x + (size_t)row * 1024))[t];
    rsum[t] = v.x + v.y + v.z + v.w;
    rsq[t]  = v.x * v.x + v.y * v.y + v.z * v.z + v.w * v.w;
    __syncthreads();
    for (int st = 128; st > 0; st >>= 1) {
        if (t < st) { rsum[t] += rsum[t + st]; rsq[t] += rsq[t + st]; }
        __syncthreads();
    }
    const float mean = rsum[0] * (1.0f / 1024.0f);
    const float var  = rsq[0] * (1.0f / 1024.0f) - mean * mean;
    const float inv  = rsqrtf(var + 1e-8f);
    const float4 g = ((const float4*)gamma)[t];
    const float4 b = ((const float4*)beta)[t];
    unsigned short* o = out + (size_t)row * 1024 + t * 4;
    o[0] = f2bf(g.x * (v.x - mean) * inv + b.x);
    o[1] = f2bf(g.y * (v.y - mean) * inv + b.y);
    o[2] = f2bf(g.z * (v.z - mean) * inv + b.z);
    o[3] = f2bf(g.w * (v.w - mean) * inv + b.w);
}

// ---------------------------------------------------------------------------
// WMMA GEMM: C[M,N] = A[M,K](bf16,row-major) * Bt[N,K](bf16, i.e. W^T)
// Tile 128x128x32, 256 threads = 8 waves, wave tile 32x64 (2x4 wmma accums).
// Double-buffered LDS tiles fed by async-to-LDS; next tile in flight while
// current tile feeds the WMMA pipe.
// EPI: 0 = scatter to q/k [B,H,S,Hd]   1 = scatter to vT [B,H,Hd,S]
//      2 = fp32 out = resid + acc + bias (attention out projection)
//      3 = bf16 out = gelu(acc + bias)  (FFN up)
//      4 = fp32 out = resid + acc + bias (FFN down, writes d_out)
// ---------------------------------------------------------------------------
#define LDT 48   // LDS row stride in ushorts (96B, multiple of 16B)

template<int EPI>
__global__ __launch_bounds__(256) void gemm_bf16_kern(
    const unsigned short* __restrict__ A, const unsigned short* __restrict__ Bt,
    const float* __restrict__ bias, const float* __restrict__ resid,
    unsigned short* __restrict__ outB, float* __restrict__ outF,
    int M, int N, int K) {
    __shared__ unsigned short ldsA[2][128 * LDT];
    __shared__ unsigned short ldsB[2][128 * LDT];

    const int t = threadIdx.x;
    const int lane = t & 31, w = t >> 5;
    const int wr = w >> 1, wc = w & 1;               // wave grid 4x2
    const int mBase = blockIdx.y * 128;
    const int nBase = blockIdx.x * 128;
    const int lr = t >> 1;                           // tile row 0..127
    const int lc = (t & 1) * 16;                     // ushort col 0 or 16
    const size_t aRow = (size_t)(mBase + lr) * K + lc;
    const size_t bRow = (size_t)(nBase + lr) * K + lc;

    unsigned aSt[2], bSt[2];
#pragma unroll
    for (int bi = 0; bi < 2; ++bi) {
        aSt[bi] = lds_off(&ldsA[bi][lr * LDT + lc]);
        bSt[bi] = lds_off(&ldsB[bi][lr * LDT + lc]);
    }

    v8f acc[2][4] = {};

    // prologue: async-load tile 0 into buffer 0
    async_load_b128(aSt[0],      A + aRow);
    async_load_b128(aSt[0] + 16, A + aRow + 8);
    async_load_b128(bSt[0],      Bt + bRow);
    async_load_b128(bSt[0] + 16, Bt + bRow + 8);
    wait_async0();
    __syncthreads();

    int cur = 0;
    for (int k0 = 0; k0 < K; k0 += 32) {
        if (k0 + 32 < K) {                           // async prefetch next tile
            const int nxt = cur ^ 1;
            async_load_b128(aSt[nxt],      A + aRow + k0 + 32);
            async_load_b128(aSt[nxt] + 16, A + aRow + k0 + 40);
            async_load_b128(bSt[nxt],      Bt + bRow + k0 + 32);
            async_load_b128(bSt[nxt] + 16, Bt + bRow + k0 + 40);
            if (k0 + 64 < K) {                       // L2 warm (global_prefetch_b8)
                __builtin_prefetch(A + aRow + k0 + 64, 0, 0);
                __builtin_prefetch(Bt + bRow + k0 + 64, 0, 0);
            }
        }

        v16bf aF[2];
#pragma unroll
        for (int i = 0; i < 2; ++i) {
            const unsigned short* rp = &ldsA[cur][(wr * 32 + i * 16 + (lane & 15)) * LDT];
            const int kb = (lane & 16) ? 8 : 0;       // A-frag K layout
            aF[i] = packfrag(*(const uint4*)(rp + kb), *(const uint4*)(rp + 16 + kb));
        }
#pragma unroll
        for (int j = 0; j < 4; ++j) {
            const unsigned short* rp = &ldsB[cur][(wc * 64 + j * 16 + (lane & 15)) * LDT];
            const int kb = (lane & 16) ? 8 : 0;       // B-frag K layout
            v16bf bF = packfrag(*(const uint4*)(rp + kb), *(const uint4*)(rp + kb + 16));
#pragma unroll
            for (int i = 0; i < 2; ++i) acc[i][j] = wmma_bf16(aF[i], bF, acc[i][j]);
        }

        // Each wave's fragment ds_loads completed before its wmma consumed them,
        // so after this barrier the old buffer is dead and the new one is full.
        wait_async0();
        __syncthreads();
        cur ^= 1;
    }

    const int half8 = (lane & 16) ? 8 : 0;            // C layout: VGPR r -> M=r / r+8
#pragma unroll
    for (int i = 0; i < 2; ++i) {
#pragma unroll
        for (int j = 0; j < 4; ++j) {
#pragma unroll
            for (int r = 0; r < 8; ++r) {
                const int row = mBase + wr * 32 + i * 16 + half8 + r;
                const int col = nBase + wc * 64 + j * 16 + (lane & 15);
                const float v = acc[i][j][r];
                if constexpr (EPI == 0 || EPI == 1) {
                    const int b = row >> 11, s = row & 2047;
                    const int h = col >> 6,  hd = col & 63;
                    const size_t o = (EPI == 0)
                        ? ((size_t)((b * 16 + h) * 2048 + s) * 64 + hd)
                        : ((size_t)((b * 16 + h) * 64 + hd) * 2048 + s);
                    outB[o] = f2bf(v);
                } else if constexpr (EPI == 2 || EPI == 4) {
                    const size_t o = (size_t)row * N + col;
                    outF[o] = resid[o] + v + bias[col];
                } else {  // EPI == 3 : GELU (tanh approx)
                    const float hh = v + bias[col];
                    const float g = 0.5f * hh *
                        (1.0f + tanhf(0.7978845608028654f * (hh + 0.044715f * hh * hh * hh)));
                    outB[(size_t)row * N + col] = f2bf(g);
                }
            }
        }
    }
}

// ---------------------------------------------------------------------------
// Flash attention, causal, Hd=64. Grid (S/128, B*H); 8 waves, 16 q-rows/wave.
// Q,K: bf16 [B,H,S,64]; Vt: bf16 [B,H,64,S]; ctx out: bf16 [B,S,D].
// K/V tiles pulled into LDS via async-to-LDS ops.
// ---------------------------------------------------------------------------
__global__ __launch_bounds__(256) void flash_attn_kern(
    const unsigned short* __restrict__ Q, const unsigned short* __restrict__ Km,
    const unsigned short* __restrict__ Vt, unsigned short* __restrict__ ctx) {
    constexpr int S = 2048, HD = 64, LDK = 80;        // 160B rows (16B aligned)
    __shared__ unsigned short ldsK[64 * LDK];
    __shared__ unsigned short ldsV[64 * LDK];
    __shared__ unsigned short ldsP[8 * 16 * LDK];

    const int t = threadIdx.x;
    const int lane = t & 31, w = t >> 5;
    const int qBase = blockIdx.x * 128;
    const int bh = blockIdx.y;
    const unsigned short* qp = Q  + (size_t)bh * S * HD;
    const unsigned short* kp = Km + (size_t)bh * S * HD;
    const unsigned short* vp = Vt + (size_t)bh * HD * S;

    // Q fragments for this wave's 16 rows (A operand, K = head-dim 0..31 / 32..63)
    v16bf aQ[2];
    {
        const unsigned short* rp = qp + (size_t)(qBase + w * 16 + (lane & 15)) * HD;
#pragma unroll
        for (int kc = 0; kc < 2; ++kc) {
            const int kb = kc * 32 + ((lane & 16) ? 8 : 0);
            aQ[kc] = packfrag(*(const uint4*)(rp + kb), *(const uint4*)(rp + kb + 16));
        }
    }

    const int half8 = (lane & 16) ? 8 : 0;
    float mrun[8], lrun[8];
#pragma unroll
    for (int r = 0; r < 8; ++r) { mrun[r] = -1e30f; lrun[r] = 0.0f; }
    v8f o[4] = {};

    const int lrow = t >> 2;              // cooperative tile load: row 0..63
    const int lcol = (t & 3) * 16;        // ushort col 0/16/32/48
    const unsigned kOff = lds_off(ldsK + lrow * LDK + lcol);
    const unsigned vOff = lds_off(ldsV + lrow * LDK + lcol);
    const int nTiles = (qBase >> 6) + 2;  // causal: keys up to qBase+127
    const float scale = 0.125f;           // 1/sqrt(64)

    for (int kt = 0; kt < nTiles; ++kt) {
        const int kBase = kt * 64;
        __syncthreads();                  // previous tile fully consumed
        {
            const unsigned short* gk = kp + (size_t)(kBase + lrow) * HD + lcol;
            const unsigned short* gv = vp + (size_t)lrow * S + kBase + lcol;
            async_load_b128(kOff,      gk);
            async_load_b128(kOff + 16, gk + 8);
            async_load_b128(vOff,      gv);
            async_load_b128(vOff + 16, gv + 8);
        }
        wait_async0();
        __syncthreads();

        // scores: Q(16x64) x K^T(64x64) -> 4 tiles of 16x16
        v8f sc[4];
#pragma unroll
        for (int j = 0; j < 4; ++j) {
            v8f s = {};
            const unsigned short* rp = ldsK + (j * 16 + (lane & 15)) * LDK;
#pragma unroll
            for (int kc = 0; kc < 2; ++kc) {
                const int kb = kc * 32 + ((lane & 16) ? 8 : 0);
                v16bf bK = packfrag(*(const uint4*)(rp + kb), *(const uint4*)(rp + kb + 16));
                s = wmma_bf16(aQ[kc], bK, s);
            }
            sc[j] = s;
        }

        // causal mask + online softmax (16-lane-half shfl reductions)
#pragma unroll
        for (int r = 0; r < 8; ++r) {
            const int qidx = qBase + w * 16 + half8 + r;
            float mx = -1e30f;
#pragma unroll
            for (int j = 0; j < 4; ++j) {
                const int kidx = kBase + j * 16 + (lane & 15);
                float sv = sc[j][r] * scale;
                sv = (kidx <= qidx) ? sv : -1e30f;
                sc[j][r] = sv;
                mx = fmaxf(mx, sv);
            }
#pragma unroll
            for (int m = 8; m >= 1; m >>= 1) mx = fmaxf(mx, __shfl_xor(mx, m, 32));
            const float mnew = fmaxf(mrun[r], mx);
            const float alpha = __expf(mrun[r] - mnew);
            mrun[r] = mnew;
            float rs = 0.0f;
#pragma unroll
            for (int j = 0; j < 4; ++j) {
                const float p = __expf(sc[j][r] - mnew);
                sc[j][r] = p;
                rs += p;
            }
#pragma unroll
            for (int m = 8; m >= 1; m >>= 1) rs += __shfl_xor(rs, m, 32);
            lrun[r] = lrun[r] * alpha + rs;
#pragma unroll
            for (int d = 0; d < 4; ++d) o[d][r] *= alpha;
        }

        // stage P (bf16) through per-wave LDS; DS ops are in-order per wave
#pragma unroll
        for (int j = 0; j < 4; ++j)
#pragma unroll
            for (int r = 0; r < 8; ++r)
                ldsP[(w * 16 + half8 + r) * LDK + j * 16 + (lane & 15)] = f2bf(sc[j][r]);

        // PV: P(16x64 keys) x V(64x64) accumulated into o
#pragma unroll
        for (int kc = 0; kc < 2; ++kc) {
            const unsigned short* rp = ldsP + (w * 16 + (lane & 15)) * LDK;
            const int kb = kc * 32 + ((lane & 16) ? 8 : 0);
            v16bf aP = packfrag(*(const uint4*)(rp + kb), *(const uint4*)(rp + kb + 16));
#pragma unroll
            for (int d = 0; d < 4; ++d) {
                const unsigned short* vr = ldsV + (d * 16 + (lane & 15)) * LDK;
                const int kb2 = kc * 32 + ((lane & 16) ? 8 : 0);
                v16bf bV = packfrag(*(const uint4*)(vr + kb2), *(const uint4*)(vr + kb2 + 16));
                o[d] = wmma_bf16(aP, bV, o[d]);
            }
        }
    }

    // finalize: ctx[B,S,D]
    const int b = bh >> 4, h = bh & 15;
#pragma unroll
    for (int r = 0; r < 8; ++r) {
        const float inv = 1.0f / lrun[r];
        const int s = qBase + w * 16 + half8 + r;
#pragma unroll
        for (int d = 0; d < 4; ++d) {
            const int dd = h * 64 + d * 16 + (lane & 15);
            ctx[(size_t)(b * S + s) * 1024 + dd] = f2bf(o[d][r] * inv);
        }
    }
}

// ---------------------------------------------------------------------------
// Host-side orchestration
// ---------------------------------------------------------------------------
extern "C" void kernel_launch(void* const* d_in, const int* in_sizes, int n_in,
                              void* d_out, int out_size, void* d_ws, size_t ws_size,
                              hipStream_t stream) {
    (void)in_sizes; (void)n_in; (void)out_size; (void)ws_size;
    const float* x    = (const float*)d_in[0];
    const float* Wq   = (const float*)d_in[1];
    const float* Wk   = (const float*)d_in[2];
    const float* Wv   = (const float*)d_in[3];
    const float* Wo   = (const float*)d_in[4];
    const float* bo   = (const float*)d_in[5];
    const float* W1   = (const float*)d_in[6];
    const float* b1   = (const float*)d_in[7];
    const float* W2   = (const float*)d_in[8];
    const float* b2   = (const float*)d_in[9];
    const float* ln1s = (const float*)d_in[10];
    const float* ln1b = (const float*)d_in[11];
    const float* ln2s = (const float*)d_in[12];
    const float* ln2b = (const float*)d_in[13];
    float* out = (float*)d_out;

    const int Bb = 4, S = 2048, D = 1024, Dff = 4096;
    const int M = Bb * S;  // 8192

    char* ws = (char*)d_ws;
    size_t off = 0;
    auto alloc = [&](size_t bytes) -> void* {
        void* p = ws + off;
        off += (bytes + 255) & ~(size_t)255;
        return p;
    };
    unsigned short* WqT  = (unsigned short*)alloc((size_t)D * D * 2);
    unsigned short* WkT  = (unsigned short*)alloc((size_t)D * D * 2);
    unsigned short* WvT  = (unsigned short*)alloc((size_t)D * D * 2);
    unsigned short* WoT  = (unsigned short*)alloc((size_t)D * D * 2);
    unsigned short* W1T  = (unsigned short*)alloc((size_t)D * Dff * 2);
    unsigned short* W2T  = (unsigned short*)alloc((size_t)Dff * D * 2);
    unsigned short* hbf  = (unsigned short*)alloc((size_t)M * D * 2);
    unsigned short* qbf  = (unsigned short*)alloc((size_t)M * D * 2);
    unsigned short* kbf  = (unsigned short*)alloc((size_t)M * D * 2);
    unsigned short* vtbf = (unsigned short*)alloc((size_t)M * D * 2);
    unsigned short* ctxb = (unsigned short*)alloc((size_t)M * D * 2);
    float*          x1   = (float*)alloc((size_t)M * D * 4);
    unsigned short* h2bf = (unsigned short*)alloc((size_t)M * D * 2);
    unsigned short* ff1b = (unsigned short*)alloc((size_t)M * Dff * 2);

    dim3 blk(256);

    // weights -> bf16 [N,K]
    wtrans_kern<<<(D * D + 255) / 256, blk, 0, stream>>>(Wq, WqT, D, D);
    wtrans_kern<<<(D * D + 255) / 256, blk, 0, stream>>>(Wk, WkT, D, D);
    wtrans_kern<<<(D * D + 255) / 256, blk, 0, stream>>>(Wv, WvT, D, D);
    wtrans_kern<<<(D * D + 255) / 256, blk, 0, stream>>>(Wo, WoT, D, D);
    wtrans_kern<<<(D * Dff + 255) / 256, blk, 0, stream>>>(W1, W1T, D, Dff);
    wtrans_kern<<<(Dff * D + 255) / 256, blk, 0, stream>>>(W2, W2T, Dff, D);

    // h = LN1(x)
    ln_kern<<<M, blk, 0, stream>>>(x, ln1s, ln1b, hbf);

    // q/k/v projections
    dim3 gD(D / 128, M / 128);
    gemm_bf16_kern<0><<<gD, blk, 0, stream>>>(hbf, WqT, nullptr, nullptr, qbf,  nullptr, M, D, D);
    gemm_bf16_kern<0><<<gD, blk, 0, stream>>>(hbf, WkT, nullptr, nullptr, kbf,  nullptr, M, D, D);
    gemm_bf16_kern<1><<<gD, blk, 0, stream>>>(hbf, WvT, nullptr, nullptr, vtbf, nullptr, M, D, D);

    // causal flash attention -> ctx [B,S,D]
    flash_attn_kern<<<dim3(S / 128, Bb * 16), blk, 0, stream>>>(qbf, kbf, vtbf, ctxb);

    // x1 = x + ctx @ Wo + bo
    gemm_bf16_kern<2><<<gD, blk, 0, stream>>>(ctxb, WoT, bo, x, nullptr, x1, M, D, D);

    // h2 = LN2(x1)
    ln_kern<<<M, blk, 0, stream>>>(x1, ln2s, ln2b, h2bf);

    // ff1 = gelu(h2 @ W1 + b1)
    gemm_bf16_kern<3><<<dim3(Dff / 128, M / 128), blk, 0, stream>>>(h2bf, W1T, b1, nullptr, ff1b, nullptr, M, Dff, D);

    // out = x1 + ff1 @ W2 + b2
    gemm_bf16_kern<4><<<gD, blk, 0, stream>>>(ff1b, W2T, b2, x1, nullptr, out, M, D, Dff);
}
